// GNN_graphpred_28750511080000
// MI455X (gfx1250) — compile-verified, matrix operand order checked by reading, not run on hardware
//
#include <hip/hip_runtime.h>
#include <math.h>

// ---------------- types ----------------
typedef __attribute__((ext_vector_type(16))) _Float16 v16h;
typedef __attribute__((ext_vector_type(8)))  _Float16 v8h;
typedef __attribute__((ext_vector_type(8)))  float    v8f;
typedef __attribute__((ext_vector_type(4)))  float    f4;
typedef unsigned int u32x4 __attribute__((ext_vector_type(4)));
typedef int          i32x8 __attribute__((ext_vector_type(8)));
typedef int          i32x4 __attribute__((ext_vector_type(4)));

#define GD 256            // feature dim D
#define GH 2              // heads
#define BM 32             // block tile M (2 waves)
#define BN 128            // block tile N (4 waves x 32)
#define BK 32             // K step (one f16 WMMA)

#if __has_builtin(__builtin_amdgcn_tensor_load_to_lds)
#define HAVE_TDM 1
#else
#define HAVE_TDM 0
#endif

__device__ __forceinline__ void wait_tensorcnt0() {
#if __has_builtin(__builtin_amdgcn_s_wait_tensorcnt)
    __builtin_amdgcn_s_wait_tensorcnt(0);
#else
    asm volatile("s_wait_tensorcnt 0x0" ::: "memory");
#endif
}

#if HAVE_TDM
// Issue one TDM 2-D tile load: BN rows x BK f16 elements, row stride K,
// from gptr (f16, row-major [n][k]) into LDS at byte offset lds_off.
__device__ __forceinline__ void tdm_load_tile(
    unsigned lds_off, const _Float16* gptr, unsigned rows_rem, unsigned K)
{
    unsigned long long ga = (unsigned long long)(size_t)gptr;
    u32x4 g0;
    g0[0] = 1u;                                   // count=1, user mode
    g0[1] = lds_off;                              // lds_addr
    g0[2] = (unsigned)ga;                         // global_addr[31:0]
    g0[3] = (unsigned)(ga >> 32) | (2u << 30);    // global_addr[56:32] | type=2
    // group1 as 4 qwords (see D# group1 bit layout)
    unsigned long long qw0 = 0x10000ull                                   // data_size=1 -> 2B
                           | ((unsigned long long)(K & 0xFFFFu) << 48);   // tensor_dim0 lo16
    unsigned long long qw1 = (unsigned long long)(K >> 16)                // tensor_dim0 hi16
                           | ((unsigned long long)rows_rem << 16)         // tensor_dim1
                           | ((unsigned long long)BK << 48);              // tile_dim0
    unsigned long long qw2 = (unsigned long long)BN                       // tile_dim1
                           | ((unsigned long long)K << 32);               // dim0_stride lo32
    unsigned long long qw3 = 0ull;                                        // stride hi, dim1_stride
    i32x8 g1;
    g1[0] = (int)qw0; g1[1] = (int)(qw0 >> 32);
    g1[2] = (int)qw1; g1[3] = (int)(qw1 >> 32);
    g1[4] = (int)qw2; g1[5] = (int)(qw2 >> 32);
    g1[6] = (int)qw3; g1[7] = (int)(qw3 >> 32);
    i32x4 z4 = {0, 0, 0, 0};
    i32x8 z8 = {0, 0, 0, 0, 0, 0, 0, 0};
    __builtin_amdgcn_tensor_load_to_lds(g0, g1, z4, z4, z8, 0);
}
#endif

// =====================================================================
// GEMM: C[M,N] = act(A[M,K] @ Bt[N,K]^T + bias[N])
// A fp32 row-major (read direct from HBM, cvt in-register).
// Bt f16, row-major [n][k] (pre-transposed weights), TDM-staged into LDS.
// 256 threads = 8 waves as 2(M) x 4(N); each wave owns 16x32 -> 2 WMMA.
// Requires K % 32 == 0.
// =====================================================================
__global__ __launch_bounds__(256) void gemm_wmma_tdm(
    const float* __restrict__ A, const _Float16* __restrict__ Bt,
    const float* __restrict__ bias, float* __restrict__ C,
    int M, int Nn, int K, int lda, int relu)
{
    __shared__ __align__(16) _Float16 sB[2][BN][BK];   // 16 KB double buffer

    const int tid  = threadIdx.x;
    const int lane = tid & 31;
    const int wid  = tid >> 5;
    const int wy   = wid >> 2;        // 0..1  (M)
    const int wx   = wid & 3;         // 0..3  (N, 32 cols each)
    const int kh   = lane >> 4;       // K-half select
    const int l15  = lane & 15;

    const int mBase = blockIdx.y * BM;
    const int nBase = blockIdx.x * BN;

    // A row for this lane's fragment (clamped; OOB rows never stored)
    int mA = mBase + wy * 16 + l15;
    if (mA > M - 1) mA = M - 1;
    const float* pa = A + (size_t)mA * lda + 8 * kh;

    v8f acc0 = {};
    v8f acc1 = {};
    int buf = 0;

#if HAVE_TDM
    const unsigned ldsB0 = (unsigned)(size_t)&sB[0][0][0];
    const unsigned ldsB1 = (unsigned)(size_t)&sB[1][0][0];
    const _Float16* btBase = Bt + (size_t)nBase * K;
    const unsigned rowsRem = (unsigned)(Nn - nBase);
    if (wid == 0) tdm_load_tile(ldsB0, btBase, rowsRem, (unsigned)K);
#endif

    for (int k0 = 0; k0 < K; k0 += BK) {
#if HAVE_TDM
        if (wid == 0) wait_tensorcnt0();
        __syncthreads();
        if (wid == 0 && (k0 + BK) < K)
            tdm_load_tile(buf ? ldsB0 : ldsB1, btBase + k0 + BK,
                          rowsRem, (unsigned)K);
#else
        __syncthreads();
        // synchronous staging: 128 rows x 4 chunks of 8 halves = 512 chunks
        for (int idx = tid; idx < (BN * BK) / 8; idx += 256) {
            int row = idx >> 2;
            int c   = (idx & 3) * 8;
            *(v8h*)&sB[buf][row][c] =
                *(const v8h*)(Bt + (size_t)(nBase + row) * K + k0 + c);
        }
        __syncthreads();
#endif
        // ---- A fragment: 16 fp32 from global, cvt to f16 in-register ----
        f4 a0 = *(const f4*)(pa + k0);
        f4 a1 = *(const f4*)(pa + k0 + 4);
        f4 a2 = *(const f4*)(pa + k0 + 16);
        f4 a3 = *(const f4*)(pa + k0 + 20);
        v16h av;
        #pragma unroll
        for (int i = 0; i < 4; i++) {
            av[i]      = (_Float16)a0[i];
            av[4 + i]  = (_Float16)a1[i];
            av[8 + i]  = (_Float16)a2[i];
            av[12 + i] = (_Float16)a3[i];
        }
        // ---- B fragments from LDS (ISA 16-bit layout) ----
        const _Float16* pb0 = &sB[buf][wx * 32 + l15][0];
        const _Float16* pb1 = &sB[buf][wx * 32 + 16 + l15][0];
        v8h b0lo = *(const v8h*)(pb0 + 8 * kh);
        v8h b0hi = *(const v8h*)(pb0 + 16 + 8 * kh);
        v8h b1lo = *(const v8h*)(pb1 + 8 * kh);
        v8h b1hi = *(const v8h*)(pb1 + 16 + 8 * kh);
        v16h bv0, bv1;
        #pragma unroll
        for (int i = 0; i < 8; i++) {
            bv0[i] = b0lo[i]; bv0[8 + i] = b0hi[i];
            bv1[i] = b1lo[i]; bv1[8 + i] = b1hi[i];
        }
        acc0 = __builtin_amdgcn_wmma_f32_16x16x32_f16(
                   false, av, false, bv0, (short)0, acc0, false, false);
        acc1 = __builtin_amdgcn_wmma_f32_16x16x32_f16(
                   false, av, false, bv1, (short)0, acc1, false, false);
        buf ^= 1;
    }

    // ---- epilogue: C layout VGPR r -> M = r + 8*kh, N = l15 ----
    #pragma unroll
    for (int t = 0; t < 2; t++) {
        int gn = nBase + wx * 32 + t * 16 + l15;
        if (gn >= Nn) continue;
        float bvv = bias ? bias[gn] : 0.0f;
        #pragma unroll
        for (int r = 0; r < 8; r++) {
            int gm = mBase + wy * 16 + (kh << 3) + r;
            if (gm < M) {
                float v = (t == 0 ? acc0[r] : acc1[r]) + bvv;
                if (relu) v = fmaxf(v, 0.0f);
                C[(size_t)gm * Nn + gn] = v;
            }
        }
    }
}

// Bt[n*K + k] = (f16) B[k*Nn + n]   (weight cvt + transpose, done once)
__global__ __launch_bounds__(256) void transpose_cvt_kernel(
    const float* __restrict__ B, _Float16* __restrict__ Bt, int K, int Nn)
{
    int t = blockIdx.x * 256 + threadIdx.x;
    if (t >= K * Nn) return;
    int n = t / K, k = t - n * K;
    Bt[t] = (_Float16)B[(size_t)k * Nn + n];
}

// =====================================================================
// h = x @ emb1_W + emb1_b + instrument @ emb2_W + emb2_b   (K=27 / K=7)
// =====================================================================
__global__ __launch_bounds__(256) void embed_kernel(
    const float* __restrict__ x, const float* __restrict__ inst,
    const float* __restrict__ W1, const float* __restrict__ b1,
    const float* __restrict__ W2, const float* __restrict__ b2,
    float* __restrict__ h, int N)
{
    int tid = blockIdx.x * 256 + threadIdx.x;
    if (tid >= N * GD) return;
    int n = tid >> 8, d = tid & (GD - 1);
    float s = b1[d] + b2[d];
    const float* xr = x + (size_t)n * 27;
    #pragma unroll
    for (int k = 0; k < 27; k++) s += xr[k] * W1[k * GD + d];
    const float* ir = inst + (size_t)n * 7;
    #pragma unroll
    for (int k = 0; k < 7; k++) s += ir[k] * W2[k * GD + d];
    h[tid] = s;
}

// ------------- order-preserving float <-> uint for atomicMax ---------
__device__ __forceinline__ unsigned mapf(float f) {
    unsigned u = __float_as_uint(f);
    return (u & 0x80000000u) ? ~u : (u | 0x80000000u);
}
__device__ __forceinline__ float unmapf(unsigned k) {
    return __uint_as_float((k & 0x80000000u) ? (k ^ 0x80000000u) : ~k);
}

// =====================================================================
// Pass A: alpha[e][h] = leaky_relu(<x_i,att_i> + <x_j+e_tab,att_j>)
// =====================================================================
__global__ __launch_bounds__(256) void edge_alpha_kernel(
    const float* __restrict__ xh, const int* __restrict__ ei,
    const int* __restrict__ ea, const float* __restrict__ att_l,
    const float* __restrict__ etab_l, float* __restrict__ alpha,
    unsigned* __restrict__ segmax, int E, int E2)
{
    int gw = blockIdx.x * 8 + (threadIdx.x >> 5);
    if (gw >= E2) return;
    int lane = threadIdx.x & 31;
    int row, col, bt;
    if (gw < E) { row = ei[gw]; col = ei[E + gw]; bt = ea[2 * gw]; }
    else        { row = col = gw - E; bt = 4; }

    const float* xi = xh + (size_t)row * (GH * GD);
    const float* xj = xh + (size_t)col * (GH * GD);
    const float* et = etab_l + bt * (GH * GD);

    float s0 = 0.f, s1 = 0.f, s2 = 0.f, s3 = 0.f;
    for (int d = lane; d < GD; d += 32) {
        s0 += xi[d]                     * att_l[d];
        s1 += (xj[d] + et[d])           * att_l[GD + d];
        s2 += xi[GD + d]                * att_l[2 * GD + d];
        s3 += (xj[GD + d] + et[GD + d]) * att_l[3 * GD + d];
    }
    #pragma unroll
    for (int m = 16; m >= 1; m >>= 1) {
        s0 += __shfl_xor(s0, m, 32);
        s1 += __shfl_xor(s1, m, 32);
        s2 += __shfl_xor(s2, m, 32);
        s3 += __shfl_xor(s3, m, 32);
    }
    if (lane == 0) {
        float a0 = s0 + s1; a0 = a0 > 0.f ? a0 : 0.2f * a0;
        float a1 = s2 + s3; a1 = a1 > 0.f ? a1 : 0.2f * a1;
        alpha[gw * 2 + 0] = a0;
        alpha[gw * 2 + 1] = a1;
        atomicMax(&segmax[row * 2 + 0], mapf(a0));
        atomicMax(&segmax[row * 2 + 1], mapf(a1));
    }
}

// Pass B: ex = exp(alpha - max[row]); den[row] += ex
__global__ __launch_bounds__(256) void edge_exp_kernel(
    const int* __restrict__ ei, const float* __restrict__ alpha,
    const unsigned* __restrict__ segmax, float* __restrict__ exb,
    float* __restrict__ den, int E, int E2)
{
    int t = blockIdx.x * 256 + threadIdx.x;
    if (t >= 2 * E2) return;
    int e = t >> 1, hh = t & 1;
    int row = (e < E) ? ei[e] : (e - E);
    float m = unmapf(segmax[row * 2 + hh]);
    float v = expf(alpha[t] - m);
    exb[t] = v;
    atomicAdd(&den[row * 2 + hh], v);
}

// Pass C: agg[row] += (x_j + e_tab) * (ex/den)
__global__ __launch_bounds__(256) void edge_scatter_kernel(
    const float* __restrict__ xh, const int* __restrict__ ei,
    const int* __restrict__ ea, const float* __restrict__ etab_l,
    const float* __restrict__ exb, const float* __restrict__ den,
    float* __restrict__ agg, int E, int E2)
{
    int gw = blockIdx.x * 8 + (threadIdx.x >> 5);
    if (gw >= E2) return;
    int lane = threadIdx.x & 31;
    int row, col, bt;
    if (gw < E) { row = ei[gw]; col = ei[E + gw]; bt = ea[2 * gw]; }
    else        { row = col = gw - E; bt = 4; }

    float a0 = exb[gw * 2 + 0] / den[row * 2 + 0];
    float a1 = exb[gw * 2 + 1] / den[row * 2 + 1];
    const float* xj = xh + (size_t)col * (GH * GD);
    const float* et = etab_l + bt * (GH * GD);
    float* o = agg + (size_t)row * (GH * GD);
    for (int d = lane; d < GD; d += 32) {
        atomicAdd(o + d,      (xj[d] + et[d]) * a0);
        atomicAdd(o + GD + d, (xj[GD + d] + et[GD + d]) * a1);
    }
}

// hn = mean over heads + gat_bias
__global__ __launch_bounds__(256) void head_mean_kernel(
    const float* __restrict__ agg, const float* __restrict__ gbias_l,
    float* __restrict__ hn, int N)
{
    int tid = blockIdx.x * 256 + threadIdx.x;
    if (tid >= N * GD) return;
    int n = tid >> 8, d = tid & (GD - 1);
    hn[tid] = 0.5f * (agg[(size_t)n * 512 + d] + agg[(size_t)n * 512 + GD + d])
              + gbias_l[d];
}

// BatchNorm over batch axis (block per feature column), optional ReLU
__global__ __launch_bounds__(256) void bn_kernel(
    const float* __restrict__ hn, const float* __restrict__ gamma_l,
    const float* __restrict__ beta_l, float* __restrict__ hout,
    int N, int relu)
{
    __shared__ float ssum[256], ssq[256];
    int d = blockIdx.x;
    int tid = threadIdx.x;
    float sum = 0.f, sq = 0.f;
    for (int i = tid; i < N; i += 256) {
        float v = hn[(size_t)i * GD + d];
        sum += v; sq += v * v;
    }
    ssum[tid] = sum; ssq[tid] = sq;
    __syncthreads();
    for (int s = 128; s > 0; s >>= 1) {
        if (tid < s) { ssum[tid] += ssum[tid + s]; ssq[tid] += ssq[tid + s]; }
        __syncthreads();
    }
    float mu  = ssum[0] / (float)N;
    float var = ssq[0] / (float)N - mu * mu;
    float scale = gamma_l[d] * rsqrtf(var + 1e-5f);
    float shift = beta_l[d] - mu * scale;
    for (int i = tid; i < N; i += 256) {
        float v = hn[(size_t)i * GD + d] * scale + shift;
        if (relu) v = fmaxf(v, 0.0f);
        hout[(size_t)i * GD + d] = v;
    }
}

// catbuf[n, 0:512] = relu(concat(h[n], t[n]))
__global__ __launch_bounds__(256) void concat_relu_kernel(
    const float* __restrict__ h, const float* __restrict__ t,
    float* __restrict__ cat, int N)
{
    int tid = blockIdx.x * 256 + threadIdx.x;
    if (tid >= N * 2 * GD) return;
    int n = tid >> 9, k = tid & (2 * GD - 1);
    float v = (k < GD) ? h[(size_t)n * GD + k] : t[(size_t)n * GD + (k - GD)];
    cat[tid] = fmaxf(v, 0.0f);
}

// =====================================================================
extern "C" void kernel_launch(void* const* d_in, const int* in_sizes, int n_in,
                              void* d_out, int out_size, void* d_ws, size_t ws_size,
                              hipStream_t stream)
{
    const float* x_in   = (const float*)d_in[0];
    const float* inst   = (const float*)d_in[1];
    const float* fp     = (const float*)d_in[2];
    const int*   ei     = (const int*)d_in[3];
    const int*   ea     = (const int*)d_in[4];
    const float* emb1_W = (const float*)d_in[5];
    const float* emb1_b = (const float*)d_in[6];
    const float* emb2_W = (const float*)d_in[7];
    const float* emb2_b = (const float*)d_in[8];
    const float* gat_W  = (const float*)d_in[9];
    const float* gat_b  = (const float*)d_in[10];
    const float* att    = (const float*)d_in[11];
    const float* gbias  = (const float*)d_in[12];
    const float* etab   = (const float*)d_in[13];
    const float* bng    = (const float*)d_in[14];
    const float* bnb    = (const float*)d_in[15];
    const float* fp_W1  = (const float*)d_in[16];
    const float* fp_b1  = (const float*)d_in[17];
    const float* fp_W2  = (const float*)d_in[18];
    const float* fp_b2  = (const float*)d_in[19];
    const float* fp_W3  = (const float*)d_in[20];
    const float* fp_b3  = (const float*)d_in[21];
    const float* cat_W  = (const float*)d_in[22];
    const float* cat_b  = (const float*)d_in[23];
    float* out = (float*)d_out;

    const int N  = in_sizes[0] / 27;
    const int E  = in_sizes[3] / 2;
    const int E2 = E + N;
    const int L  = 5;

    // ---------------- workspace layout ----------------
    char* ws = (char*)d_ws;
    size_t off = 0;
    float* h    = (float*)(ws + off); off += (size_t)N * GD * 4;
    float* hn   = (float*)(ws + off); off += (size_t)N * GD * 4;       // fp bufA
    float* xh   = (float*)(ws + off); off += (size_t)N * GH * GD * 4;  // fp bufB
    float* agg  = (float*)(ws + off); off += (size_t)N * GH * GD * 4;  // catbuf
    float* alpha = (float*)(ws + off); off += (size_t)E2 * GH * 4;
    float* exb   = (float*)(ws + off); off += (size_t)E2 * GH * 4;
    float* den   = (float*)(ws + off); off += (size_t)N * GH * 4;
    unsigned* segmax = (unsigned*)(ws + off); off += (size_t)N * GH * 4;
    // f16 transposed weights
    _Float16* gatWt = (_Float16*)(ws + off); off += (size_t)L * GD * GH * GD * 2;
    _Float16* fpW1t = (_Float16*)(ws + off); off += (size_t)4096 * GD * 2;
    _Float16* fpW2t = (_Float16*)(ws + off); off += (size_t)GD * GD * 2;
    _Float16* fpW3t = (_Float16*)(ws + off); off += (size_t)GD * GD * 2;
    _Float16* catWt = (_Float16*)(ws + off); off += (size_t)2 * GD * GD * 2;

    const int edgeBlocks = (E2 + 7) / 8;
    const int gyM = (N + BM - 1) / BM;

    // ---------------- one-time weight cvt+transpose ----------------
    for (int l = 0; l < L; l++)
        transpose_cvt_kernel<<<(GD * GH * GD + 255) / 256, 256, 0, stream>>>(
            gat_W + (size_t)l * GD * GH * GD,
            gatWt + (size_t)l * GD * GH * GD, GD, GH * GD);
    transpose_cvt_kernel<<<(4096 * GD + 255) / 256, 256, 0, stream>>>(
        fp_W1, fpW1t, 4096, GD);
    transpose_cvt_kernel<<<(GD * GD + 255) / 256, 256, 0, stream>>>(
        fp_W2, fpW2t, GD, GD);
    transpose_cvt_kernel<<<(GD * GD + 255) / 256, 256, 0, stream>>>(
        fp_W3, fpW3t, GD, GD);
    transpose_cvt_kernel<<<(2 * GD * GD + 255) / 256, 256, 0, stream>>>(
        cat_W, catWt, 2 * GD, GD);

    // ---------------- input embeddings ----------------
    embed_kernel<<<(N * GD + 255) / 256, 256, 0, stream>>>(
        x_in, inst, emb1_W, emb1_b, emb2_W, emb2_b, h, N);

    // ---------------- GAT layers ----------------
    for (int l = 0; l < L; l++) {
        const _Float16* Wt  = gatWt + (size_t)l * GD * GH * GD;
        const float*    bl  = gat_b + (size_t)l * (GH * GD);
        const float*    atl = att + (size_t)l * GH * 2 * GD;
        const float*    etl = etab + (size_t)l * 5 * (GH * GD);

        // xh = h @ W + b   (M=N, N=512, K=256)
        {
            dim3 grid((GH * GD) / BN, gyM);
            gemm_wmma_tdm<<<grid, 256, 0, stream>>>(
                h, Wt, bl, xh, N, GH * GD, GD, GD, 0);
        }
        (void)hipMemsetAsync(segmax, 0, (size_t)N * GH * 4, stream);
        (void)hipMemsetAsync(den,    0, (size_t)N * GH * 4, stream);
        (void)hipMemsetAsync(agg,    0, (size_t)N * GH * GD * 4, stream);

        edge_alpha_kernel<<<edgeBlocks, 256, 0, stream>>>(
            xh, ei, ea, atl, etl, alpha, segmax, E, E2);
        edge_exp_kernel<<<(2 * E2 + 255) / 256, 256, 0, stream>>>(
            ei, alpha, segmax, exb, den, E, E2);
        edge_scatter_kernel<<<edgeBlocks, 256, 0, stream>>>(
            xh, ei, ea, etl, exb, den, agg, E, E2);
        head_mean_kernel<<<(N * GD + 255) / 256, 256, 0, stream>>>(
            agg, gbias + (size_t)l * GD, hn, N);
        bn_kernel<<<GD, 256, 0, stream>>>(
            hn, bng + (size_t)l * GD, bnb + (size_t)l * GD, h, N, (l < L - 1) ? 1 : 0);
    }

    // ---------------- fingerprint MLP ----------------
    float* bufA = hn;
    float* bufB = xh;
    {
        dim3 grid(GD / BN, gyM);
        gemm_wmma_tdm<<<grid, 256, 0, stream>>>(
            fp, fpW1t, fp_b1, bufA, N, GD, 4096, 4096, 0);
        gemm_wmma_tdm<<<grid, 256, 0, stream>>>(
            bufA, fpW2t, fp_b2, bufB, N, GD, GD, GD, 1);
        gemm_wmma_tdm<<<grid, 256, 0, stream>>>(
            bufB, fpW3t, fp_b3, bufA, N, GD, GD, GD, 0);
    }

    // ---------------- concat + head ----------------
    float* catbuf = agg;
    concat_relu_kernel<<<(N * 2 * GD + 255) / 256, 256, 0, stream>>>(
        h, bufA, catbuf, N);
    {
        dim3 grid(GD / BN, gyM);
        gemm_wmma_tdm<<<grid, 256, 0, stream>>>(
            catbuf, catWt, cat_b, out, N, GD, 2 * GD, 2 * GD, 0);
    }
}
